// CausalSelfAttention_87110526697958
// MI455X (gfx1250) — compile-verified, matrix operand order checked by reading, not run on hardware
//
#include <hip/hip_runtime.h>

// CausalSelfAttention for MI455X (gfx1250, wave32, WMMA + TDM).
// B=2 T=2048 C=1024 H=16 D=64. All matmuls via v_wmma_f32_16x16x32_bf16.
// bf16 tiles staged into LDS by the Tensor Data Mover when available.

#define BATCH 2
#define HEADS 16
#define SEQ   2048
#define HDIM  64
#define CDIM  1024
#define NQKV  3072

typedef __attribute__((ext_vector_type(16))) __bf16 v16bf;
typedef __attribute__((ext_vector_type(8)))  float  v8f;

// ---- feature detection -----------------------------------------------------
#if __has_builtin(__builtin_amdgcn_tensor_load_to_lds) && \
    __has_builtin(__builtin_amdgcn_s_wait_tensorcnt)
#define HAVE_TDM 1
#else
#define HAVE_TDM 0
#endif
#if __has_builtin(__builtin_amdgcn_global_load_async_to_lds_b128)
#define HAVE_ASYNC 1
#else
#define HAVE_ASYNC 0
#endif

typedef __attribute__((ext_vector_type(4))) unsigned int tdm_u4;
typedef __attribute__((ext_vector_type(8))) int          tdm_i8;
typedef __attribute__((ext_vector_type(4))) int          tdm_i4;
typedef __attribute__((ext_vector_type(4))) int          asy_i4;

static __device__ __forceinline__ unsigned lds_off(const void* p) {
  // LDS byte offset of a __shared__ object (addrspacecast generic->local).
  return (unsigned)(unsigned long long)
      (__attribute__((address_space(3))) const char*)p;
}

// Build the loop-invariant D# group1: tile_w x tile_h bf16 elements, global
// row stride row_stride (elements), packed contiguously into LDS.
static __device__ __forceinline__ tdm_i8 tdm_make_g1(unsigned tile_w,
                                                     unsigned tile_h,
                                                     unsigned long long row_stride) {
  union { tdm_i8 v; unsigned long long q[4]; } g1;
  g1.q[0] = (1ull << 16)                                       // data_size=2B
          | ((unsigned long long)tile_w << 48);                // tensor_dim0[15:0]
  g1.q[1] = ((unsigned long long)tile_w >> 16)                 // tensor_dim0[31:16]
          | ((unsigned long long)tile_h << 16)                 // tensor_dim1
          | ((unsigned long long)tile_w << 48);                // tile_dim0
  g1.q[2] = (unsigned long long)tile_h                         // tile_dim1
          | ((row_stride & 0xffffffffull) << 32);              // dim0_stride[31:0]
  g1.q[3] = (row_stride >> 32) & 0xffffull;                    // dim0_stride[47:32]
  return g1.v;
}

// Issue one TDM tile load: only D# group0 (addresses) varies per call.
static __device__ __forceinline__ void tdm_issue(const void* gptr, unsigned ldsoff,
                                                 tdm_i8 g1) {
#if HAVE_TDM
  const unsigned long long ga = (unsigned long long)gptr;
  union { tdm_u4 v; unsigned u[4]; } g0;
  g0.u[0] = 1u;                                      // count=1, user mode
  g0.u[1] = ldsoff;                                  // lds_addr
  g0.u[2] = (unsigned)ga;                            // global_addr[31:0]
  g0.u[3] = (unsigned)((ga >> 32) & 0x01ffffffu)     // global_addr[56:32]
          | (2u << 30);                              // type=2 ("image")
  tdm_i4 z4; z4[0] = z4[1] = z4[2] = z4[3] = 0;
#if __clang_major__ >= 23
  tdm_i8 z8;
#pragma unroll
  for (int i = 0; i < 8; ++i) z8[i] = 0;
  __builtin_amdgcn_tensor_load_to_lds(g0.v, g1, z4, z4, z8, 0);
#else
  __builtin_amdgcn_tensor_load_to_lds(g0.v, g1, z4, z4, 0);
#endif
#else
  (void)gptr; (void)ldsoff; (void)g1;
#endif
}

static __device__ __forceinline__ void tdm_wait() {
#if HAVE_TDM
  __builtin_amdgcn_s_wait_tensorcnt(0);
#endif
}

// 16-byte global->LDS stage (async-to-LDS when available).
static __device__ __forceinline__ void stage16(void* l, const void* g) {
#if HAVE_ASYNC
  __builtin_amdgcn_global_load_async_to_lds_b128(
      (__attribute__((address_space(1))) asy_i4*)(unsigned long long)g,
      (__attribute__((address_space(3))) asy_i4*)l, 0, 0);
#else
  *(uint4*)l = *(const uint4*)g;
#endif
}
static __device__ __forceinline__ void stage_wait() {
#if HAVE_ASYNC
#if __has_builtin(__builtin_amdgcn_s_wait_asynccnt)
  __builtin_amdgcn_s_wait_asynccnt(0);
#else
  asm volatile("s_wait_asynccnt 0x0" ::: "memory");
#endif
#endif
}

static __device__ __forceinline__ v8f vzero() {
  v8f z;
#pragma unroll
  for (int i = 0; i < 8; ++i) z[i] = 0.0f;
  return z;
}

// Load a 16x32 bf16 A/B fragment from LDS laid out row-major [16][stride].
// CDNA5 16-bit layout: lanes 0-15 -> row lane, K {0..7,16..23}; lanes 16-31
// same rows, K {8..15,24..31}; VGPR j packs one K pair.
static __device__ __forceinline__ v16bf load_frag(const __bf16* __restrict__ base,
                                                  int stride, int lane) {
  union { v16bf v; unsigned int u[8]; } f;
  const __bf16* rowp = base + (lane & 15) * stride;
  const int kh = (lane >> 4) << 3;
#pragma unroll
  for (int j = 0; j < 8; ++j) {
    const int k = ((j < 4) ? (j << 1) : (16 + ((j - 4) << 1))) + kh;
    f.u[j] = *(const unsigned int*)(rowp + k);
  }
  return f.v;
}

static __device__ __forceinline__ v8f wmma_bf16(v16bf a, v16bf b, v8f c) {
  return __builtin_amdgcn_wmma_f32_16x16x32_bf16(false, a, false, b, (short)0, c,
                                                 false, false);
}

// ---------------------------------------------------------------------------
// Kernel 1: x[4096,1024] @ w_qkv[1024,3072] -> Q/K bf16 [B,H,T,D], V bf16
// stored TRANSPOSED [B,H,D,T] so attention V staging is a plain 2-D copy.
// ---------------------------------------------------------------------------
__global__ __launch_bounds__(256)
void qkv_gemm_kernel(const float* __restrict__ x, const float* __restrict__ w,
                     __bf16* __restrict__ Q, __bf16* __restrict__ K,
                     __bf16* __restrict__ V) {
  __shared__ __align__(16) __bf16 sA[128 * 32];  // [m][k]
  __shared__ __align__(16) __bf16 sB[128 * 32];  // [n][k] (transposed)
  const int tid = threadIdx.x;
  const int lane = tid & 31, wave = tid >> 5;
  const int wm = wave & 3, wn = wave >> 2;
  const int m0 = blockIdx.x * 128, n0 = blockIdx.y * 128;

  v8f acc[2][4];
#pragma unroll
  for (int i = 0; i < 2; ++i)
#pragma unroll
    for (int j = 0; j < 4; ++j) acc[i][j] = vzero();

  for (int k0 = 0; k0 < CDIM; k0 += 32) {
#pragma unroll
    for (int j = 0; j < 4; ++j) {
      const int idx = tid + 256 * j;
      const int row = idx >> 3, kq = (idx & 7) << 2;
      const float4 f = *(const float4*)(x + (size_t)(m0 + row) * CDIM + k0 + kq);
      sA[row * 32 + kq + 0] = (__bf16)f.x;
      sA[row * 32 + kq + 1] = (__bf16)f.y;
      sA[row * 32 + kq + 2] = (__bf16)f.z;
      sA[row * 32 + kq + 3] = (__bf16)f.w;
    }
#pragma unroll
    for (int j = 0; j < 4; ++j) {
      const int idx = tid + 256 * j;
      const int kr = idx >> 5, n = (idx & 31) << 2;
      const float4 f = *(const float4*)(w + (size_t)(k0 + kr) * NQKV + n0 + n);
      sB[(n + 0) * 32 + kr] = (__bf16)f.x;
      sB[(n + 1) * 32 + kr] = (__bf16)f.y;
      sB[(n + 2) * 32 + kr] = (__bf16)f.z;
      sB[(n + 3) * 32 + kr] = (__bf16)f.w;
    }
    __syncthreads();
#pragma unroll
    for (int mt = 0; mt < 2; ++mt) {
      const v16bf a = load_frag(sA + (32 * wm + 16 * mt) * 32, 32, lane);
#pragma unroll
      for (int nt = 0; nt < 4; ++nt) {
        const v16bf b = load_frag(sB + (64 * wn + 16 * nt) * 32, 32, lane);
        acc[mt][nt] = wmma_bf16(a, b, acc[mt][nt]);
      }
    }
    __syncthreads();
  }

  // C-frag mapping: VGPR r, lane l -> (row r + 8*(l>>4), col l&15).
  // Branchless epilogue: base pointer and offset chosen with selects so the
  // compiler emits v_cndmask instead of exec-mask branch ladders.
  const int cl = lane & 15, hi = lane >> 4;
#pragma unroll
  for (int mt = 0; mt < 2; ++mt)
#pragma unroll
    for (int nt = 0; nt < 4; ++nt)
#pragma unroll
      for (int r = 0; r < 8; ++r) {
        const int m = m0 + 32 * wm + 16 * mt + r + 8 * hi;
        const int n = n0 + 64 * wn + 16 * nt + cl;
        const int bb = m >> 11, t = m & (SEQ - 1);
        const int which = n >> 10, n2 = n & 1023;
        const int h = n2 >> 6, d = n2 & 63;
        const size_t bh = (size_t)(bb * HEADS + h);
        const size_t off_qk = (bh * SEQ + t) * HDIM + d;
        const size_t off_v  = (bh * HDIM + d) * SEQ + t;  // transposed V layout
        __bf16* dst = (which == 0) ? Q : (which == 1) ? K : V;
        const size_t off = (which == 2) ? off_v : off_qk;
        dst[off] = (__bf16)acc[mt][nt][r];
      }
}

// ---------------------------------------------------------------------------
// Kernel 2: flash attention with sink column. 128-query tile per block; one
// wave owns 16 query rows; stream 64-key blocks with online softmax. Q/K/Vt
// tiles staged by the Tensor Data Mover (fallback: async/plain copies).
// ---------------------------------------------------------------------------
__global__ __launch_bounds__(256)
void flash_attn_kernel(const __bf16* __restrict__ Q, const __bf16* __restrict__ Kg,
                       const __bf16* __restrict__ Vt, const float* __restrict__ sink,
                       __bf16* __restrict__ Y) {
  __shared__ __align__(16) __bf16 sQ[128 * 64];     // [tq][d]
  __shared__ __align__(16) __bf16 sK[64 * 64];      // [tk][d] == B-frag for Q.K^T
  __shared__ __align__(16) __bf16 sVt[64 * 64];     // [d][tk] == B-frag for P.V
  __shared__ __align__(16) __bf16 sP[8][16 * 64];   // per-wave probs [row][tk]

  const int tid = threadIdx.x;
  const int lane = tid & 31, wave = tid >> 5;
  const int cl = lane & 15, hi = lane >> 4;
  const int qb = blockIdx.x * 128;
  const int bh = blockIdx.y;
  const int b = bh >> 4, h = bh & 15;
  const size_t base = (size_t)bh * SEQ * HDIM;  // same slab size for [T,D] and [D,T]

#if HAVE_TDM
  // Loop-invariant descriptors (compile-time constants -> immediates).
  const tdm_i8 g1_q  = tdm_make_g1(HDIM, 128, HDIM);
  const tdm_i8 g1_k  = tdm_make_g1(HDIM, 64, HDIM);
  const tdm_i8 g1_vt = tdm_make_g1(64, HDIM, SEQ);
  if (wave == 0) {
    tdm_issue(Q + base + (size_t)qb * HDIM, lds_off(sQ), g1_q);
    tdm_wait();
  }
#else
  {
    const __bf16* gq = Q + base + (size_t)qb * HDIM;
#pragma unroll
    for (int j = 0; j < 4; ++j) {
      const int idx = tid + 256 * j;           // 1024 x 16B
      stage16(sQ + idx * 8, gq + idx * 8);
    }
    stage_wait();
  }
#endif

  // online softmax state, 8 rows per lane (rows r + 8*hi of this wave's tile)
  float m_run[8], s_run[8];
  const float slog = sink[h];  // sink column: init m = sink_logit, s = 1, y = 0
#pragma unroll
  for (int r = 0; r < 8; ++r) { m_run[r] = slog; s_run[r] = 1.0f; }
  v8f y[4];
#pragma unroll
  for (int nt = 0; nt < 4; ++nt) y[nt] = vzero();

  const int kb_max = (qb >> 6) + 1;  // causal: keys <= qb+127
  for (int kb = 0; kb <= kb_max; ++kb) {
    __syncthreads();  // all waves done reading previous sK/sVt (and sQ staged)
#if HAVE_TDM
    if (wave == 0) {
      tdm_issue(Kg + base + (size_t)kb * 64 * HDIM, lds_off(sK), g1_k);
      tdm_issue(Vt + base + (size_t)kb * 64, lds_off(sVt), g1_vt);
      tdm_wait();
    }
#else
    {
      const __bf16* gk = Kg + base + (size_t)kb * 64 * HDIM;
      const __bf16* gv = Vt + base + (size_t)kb * 64;
#pragma unroll
      for (int j = 0; j < 2; ++j) {
        const int idx = tid + 256 * j;  // 512 x 16B each
        stage16(sK + idx * 8, gk + idx * 8);
        const int d = idx >> 2, seg = (idx & 3) << 3;
        stage16(sVt + d * 64 + seg, gv + (size_t)d * SEQ + seg);
      }
      stage_wait();
    }
#endif
    if (kb < kb_max)  // warm L2/WGP$ for the next key block
      __builtin_prefetch(Kg + base + (size_t)(kb + 1) * 64 * HDIM + tid * 8, 0, 1);
    __syncthreads();

    // scores: (16 q-rows) x (64 keys), contract over d=64 in two K=32 steps
    v8f sc[4];
#pragma unroll
    for (int nt = 0; nt < 4; ++nt) sc[nt] = vzero();
#pragma unroll
    for (int kc = 0; kc < 2; ++kc) {
      const v16bf a = load_frag(sQ + (16 * wave) * 64 + 32 * kc, 64, lane);
#pragma unroll
      for (int nt = 0; nt < 4; ++nt) {
        const v16bf bf = load_frag(sK + (16 * nt) * 64 + 32 * kc, 64, lane);
        sc[nt] = wmma_bf16(a, bf, sc[nt]);
      }
    }

    // scale + causal mask + block row-max
    float pmax[8];
#pragma unroll
    for (int r = 0; r < 8; ++r) pmax[r] = -3.0e38f;
#pragma unroll
    for (int nt = 0; nt < 4; ++nt)
#pragma unroll
      for (int r = 0; r < 8; ++r) {
        const int tq = qb + 16 * wave + r + 8 * hi;
        const int tk = kb * 64 + 16 * nt + cl;
        float v = sc[nt][r] * 0.125f;  // 1/sqrt(64)
        v = (tk <= tq) ? v : -3.0e38f;
        sc[nt][r] = v;
        pmax[r] = fmaxf(pmax[r], v);
      }
    // rows live in 16-lane halves; xor masks 1,2,4,8 stay within a half
#pragma unroll
    for (int r = 0; r < 8; ++r)
#pragma unroll
      for (int off = 1; off < 16; off <<= 1)
        pmax[r] = fmaxf(pmax[r], __shfl_xor(pmax[r], off, 32));

    float scale[8], rsum[8];
#pragma unroll
    for (int r = 0; r < 8; ++r) {
      const float mnew = fmaxf(m_run[r], pmax[r]);
      scale[r] = __expf(m_run[r] - mnew);
      m_run[r] = mnew;
      rsum[r] = 0.0f;
    }
#pragma unroll
    for (int nt = 0; nt < 4; ++nt)
#pragma unroll
      for (int r = 0; r < 8; ++r) {
        const float p = __expf(sc[nt][r] - m_run[r]);
        sc[nt][r] = p;
        rsum[r] += p;
      }
#pragma unroll
    for (int r = 0; r < 8; ++r) {
#pragma unroll
      for (int off = 1; off < 16; off <<= 1) rsum[r] += __shfl_xor(rsum[r], off, 32);
      s_run[r] = s_run[r] * scale[r] + rsum[r];
    }
#pragma unroll
    for (int nt = 0; nt < 4; ++nt)
#pragma unroll
      for (int r = 0; r < 8; ++r) y[nt][r] *= scale[r];

    // C-layout -> A-layout transpose through this wave's private LDS slab
#pragma unroll
    for (int nt = 0; nt < 4; ++nt)
#pragma unroll
      for (int r = 0; r < 8; ++r)
        sP[wave][(r + 8 * hi) * 64 + 16 * nt + cl] = (__bf16)sc[nt][r];

    // DS ops are in-order per wave; drain the split DS counter before re-read
    asm volatile("s_wait_dscnt 0" ::: "memory");

    // y += P(16x64) @ V(64x64)
#pragma unroll
    for (int kc = 0; kc < 2; ++kc) {
      const v16bf a = load_frag(&sP[wave][0] + 32 * kc, 64, lane);
#pragma unroll
      for (int nt = 0; nt < 4; ++nt) {
        const v16bf bf = load_frag(sVt + (16 * nt) * 64 + 32 * kc, 64, lane);
        y[nt] = wmma_bf16(a, bf, y[nt]);
      }
    }
  }

#pragma unroll
  for (int r = 0; r < 8; ++r) s_run[r] = 1.0f / s_run[r];
#pragma unroll
  for (int nt = 0; nt < 4; ++nt)
#pragma unroll
    for (int r = 0; r < 8; ++r) {
      const int tq = qb + 16 * wave + r + 8 * hi;
      const int d = 16 * nt + cl;
      Y[((size_t)b * SEQ + tq) * (HEADS * HDIM) + h * HDIM + d] =
          (__bf16)(y[nt][r] * s_run[r]);
    }
}

// ---------------------------------------------------------------------------
// Kernel 3: Y[4096,1024] (bf16) @ w_proj[1024,1024] (f32->bf16) -> out f32
// ---------------------------------------------------------------------------
__global__ __launch_bounds__(256)
void proj_gemm_kernel(const __bf16* __restrict__ Y, const float* __restrict__ w,
                      float* __restrict__ out) {
  __shared__ __align__(16) __bf16 sA[128 * 32];
  __shared__ __align__(16) __bf16 sB[128 * 32];
  const int tid = threadIdx.x;
  const int lane = tid & 31, wave = tid >> 5;
  const int wm = wave & 3, wn = wave >> 2;
  const int m0 = blockIdx.x * 128, n0 = blockIdx.y * 128;

  v8f acc[2][4];
#pragma unroll
  for (int i = 0; i < 2; ++i)
#pragma unroll
    for (int j = 0; j < 4; ++j) acc[i][j] = vzero();

  for (int k0 = 0; k0 < CDIM; k0 += 32) {
#pragma unroll
    for (int j = 0; j < 2; ++j) {
      const int idx = tid + 256 * j;  // 512 x 16B = 4096 bf16
      const int row = idx >> 2, seg = (idx & 3) << 3;
      stage16(sA + row * 32 + seg, Y + (size_t)(m0 + row) * CDIM + k0 + seg);
    }
#pragma unroll
    for (int j = 0; j < 4; ++j) {
      const int idx = tid + 256 * j;
      const int kr = idx >> 5, n = (idx & 31) << 2;
      const float4 f = *(const float4*)(w + (size_t)(k0 + kr) * CDIM + n0 + n);
      sB[(n + 0) * 32 + kr] = (__bf16)f.x;
      sB[(n + 1) * 32 + kr] = (__bf16)f.y;
      sB[(n + 2) * 32 + kr] = (__bf16)f.z;
      sB[(n + 3) * 32 + kr] = (__bf16)f.w;
    }
    stage_wait();
    __syncthreads();
#pragma unroll
    for (int mt = 0; mt < 2; ++mt) {
      const v16bf a = load_frag(sA + (32 * wm + 16 * mt) * 32, 32, lane);
#pragma unroll
      for (int nt = 0; nt < 4; ++nt) {
        const v16bf b = load_frag(sB + (64 * wn + 16 * nt) * 32, 32, lane);
        acc[mt][nt] = wmma_bf16(a, b, acc[mt][nt]);
      }
    }
    __syncthreads();
  }

  const int cl = lane & 15, hi = lane >> 4;
#pragma unroll
  for (int mt = 0; mt < 2; ++mt)
#pragma unroll
    for (int nt = 0; nt < 4; ++nt)
#pragma unroll
      for (int r = 0; r < 8; ++r) {
        const int m = m0 + 32 * wm + 16 * mt + r + 8 * hi;
        const int n = n0 + 64 * wn + 16 * nt + cl;
        out[(size_t)m * CDIM + n] = acc[mt][nt][r];
      }
}

// ---------------------------------------------------------------------------
extern "C" void kernel_launch(void* const* d_in, const int* in_sizes, int n_in,
                              void* d_out, int out_size, void* d_ws, size_t ws_size,
                              hipStream_t stream) {
  (void)in_sizes; (void)n_in; (void)out_size; (void)ws_size;
  const float* x      = (const float*)d_in[0];
  const float* w_qkv  = (const float*)d_in[1];
  const float* w_proj = (const float*)d_in[2];
  const float* sink   = (const float*)d_in[3];
  float* out = (float*)d_out;

  const size_t per = (size_t)BATCH * HEADS * SEQ * HDIM;  // 4 Mi elements
  __bf16* Q  = (__bf16*)d_ws;
  __bf16* K  = Q + per;
  __bf16* Vt = K + per;
  __bf16* Y  = Vt + per;

  qkv_gemm_kernel<<<dim3((BATCH * SEQ) / 128, NQKV / 128), 256, 0, stream>>>(
      x, w_qkv, Q, K, Vt);
  flash_attn_kernel<<<dim3(SEQ / 128, BATCH * HEADS), 256, 0, stream>>>(
      Q, K, Vt, sink, Y);
  proj_gemm_kernel<<<dim3((BATCH * SEQ) / 128, CDIM / 128), 256, 0, stream>>>(
      Y, w_proj, out);
}
